// SDF_54803782697191
// MI455X (gfx1250) — matrix operand order repaired and checked by reading, not compiled, not run on hardware
//
#include <hip/hip_runtime.h>

typedef __attribute__((ext_vector_type(2)))  _Float16 v2h;
typedef __attribute__((ext_vector_type(8)))  _Float16 v8h;
typedef __attribute__((ext_vector_type(16))) _Float16 v16h;
typedef __attribute__((ext_vector_type(8)))  float    v8f;

#define LOG2_T 19
#define TSIZE  (1 << LOG2_T)
#define P1 2654435761u
#define P2 805459861u

// scale_l = 16 * 2^(7l/15) - 1 ; res_l = ceil(scale_l)+1 ; dense iff res^3 <= 2^19 (levels 0..4)
__constant__ float c_scale[16] = {
    15.0f,        21.1106067f,  29.5549306f,  41.2242527f,
    57.3503890f,  79.6349466f,  110.4304716f, 152.9872434f,
    211.8095878f, 293.0667843f, 405.3746708f, 560.5638083f,
    775.0468757f, 1071.4548454f, 1481.1116419f, 2047.0f};
__constant__ int c_res[16] = {16, 23, 31, 43, 59, 81, 112, 154,
                              213, 295, 407, 562, 777, 1073, 1483, 2048};

// softplus(10z)/10 = max(z,0) + log2(1 + exp2(-10*log2e*|z|)) * (ln2/10)
// Raw v_exp_f32 / v_log_f32: log2 arg is in (1,2], so no edge-case fixup needed.
__device__ __forceinline__ float softplus10(float z) {
  float e  = __builtin_amdgcn_exp2f(-14.4269504088896f * fabsf(z));
  float lg = __builtin_amdgcn_logf(1.0f + e);
  return fmaf(lg, 0.0693147180559945f, fmaxf(z, 0.0f));
}

__device__ __forceinline__ v16h cat8(v8h lo, v8h hi) {
  return __builtin_shufflevector(lo, hi, 0, 1, 2, 3, 4, 5, 6, 7,
                                 8, 9, 10, 11, 12, 13, 14, 15);
}

__launch_bounds__(256)
__global__ void ngp_sdf_kernel(const float* __restrict__ x,
                               const float* __restrict__ table,
                               const float* __restrict__ w1,
                               const float* __restrict__ w2,
                               const float* __restrict__ w3,
                               const float* __restrict__ w4,
                               float* __restrict__ out, int n_tiles) {
  // Weight B-fragments (shared by all waves of the block): lane = column n (&15),
  // element e <-> K = (e&7) + 8*(lane>=16) + 16*(e>=8), per 16-bit B-matrix 32x16 layout.
  __shared__ __attribute__((aligned(64))) _Float16 sB1[4][32][16];      //  4 KB
  __shared__ __attribute__((aligned(64))) _Float16 sB2[2][4][32][16];   //  8 KB
  __shared__ __attribute__((aligned(64))) _Float16 sB3[2][4][32][16];   //  8 KB
  __shared__ __attribute__((aligned(64))) float    sW4[64];             // 256 B
  // Per-wave staging: encoding (A-matrix 16x32) and hidden activations (16x64).
  __shared__ __attribute__((aligned(64))) _Float16 sEnc[8][16][32];     //  8 KB
  __shared__ __attribute__((aligned(64))) _Float16 sH[8][16][64];       // 16 KB

  const int tid  = threadIdx.x;
  const int lane = tid & 31;
  const int wv   = tid >> 5;
  const int m    = lane & 15;      // row (point) within 16-point tile
  const int hi16 = lane >> 4;      // 0: lanes 0-15, 1: lanes 16-31
  const int k0   = hi16 * 8;       // A-matrix K base per documented layout

  // ---- pre-pack weights f32 -> f16 WMMA B-fragments in LDS (once per block)
  for (int i = tid; i < 4 * 32 * 16; i += 256) {
    int e = i & 15, ln = (i >> 4) & 31, t = i >> 9;
    int n = t * 16 + (ln & 15);
    int k = (e & 7) + ((ln & 16) ? 8 : 0) + ((e & 8) ? 16 : 0);
    sB1[t][ln][e] = (_Float16)w1[n * 32 + k];
  }
  for (int i = tid; i < 2 * 4 * 32 * 16; i += 256) {
    int e = i & 15, ln = (i >> 4) & 31, t = (i >> 9) & 3, kc = i >> 11;
    int n = t * 16 + (ln & 15);
    int k = kc * 32 + (e & 7) + ((ln & 16) ? 8 : 0) + ((e & 8) ? 16 : 0);
    sB2[kc][t][ln][e] = (_Float16)w2[n * 64 + k];
    sB3[kc][t][ln][e] = (_Float16)w3[n * 64 + k];
  }
  if (tid < 64) sW4[tid] = w4[tid];
  __syncthreads();

  const int wavesTotal = (gridDim.x * blockDim.x) >> 5;
  const int gw = (blockIdx.x * blockDim.x + tid) >> 5;

  for (int tile = gw; tile < n_tiles; tile += wavesTotal) {
    // ================= hash-grid encoding =================
    // Even/odd level split: lanes 0-15 do levels {0,2,..,14}, lanes 16-31 do
    // {1,3,..,15} of point m -> the dense/hash branch is wave-uniform for 7 of
    // 8 unrolled iterations (only the level 4/5 pair mixes paths).
    const int p = tile * 16 + m;
    const float px = x[3 * p + 0];
    const float py = x[3 * p + 1];
    const float pz = x[3 * p + 2];
#pragma unroll
    for (int ll = 0; ll < 8; ++ll) {
      const int l = 2 * ll + hi16;
      const float scale = c_scale[l];
      const int res = c_res[l];
      const bool dense = (res <= 80);  // res^3 <= 2^19
      float fx = fmaf(px, scale, 0.5f);
      float fy = fmaf(py, scale, 0.5f);
      float fz = fmaf(pz, scale, 0.5f);
      float gx = floorf(fx), gy = floorf(fy), gz = floorf(fz);
      float wx = fx - gx, wy = fy - gy, wz = fz - gz;
      const int ix = (int)gx, iy = (int)gy, iz = (int)gz;
      // trilinear corner weights: 4 xy-products + per-corner z scale
      const float ux0 = 1.0f - wx, uy0 = 1.0f - wy;
      const float wxy[4] = {ux0 * uy0, wx * uy0, ux0 * wy, wx * wy};
      const float uzz[2] = {1.0f - wz, wz};
      const float* tab = table + ((size_t)l << (LOG2_T + 1));
      float f0 = 0.0f, f1 = 0.0f;
      if (dense) {
        // x in [0,1) => fx >= 0.5 => ix >= 0: lower clamp is dead, upper only.
        const int r1 = res - 1;
        const int qx[2] = {min(ix, r1), min(ix + 1, r1)};
        const int qy[2] = {min(iy, r1) * res, min(iy + 1, r1) * res};
        const int rr = res * res;
        const int qz[2] = {min(iz, r1) * rr, min(iz + 1, r1) * rr};
#pragma unroll
        for (int c = 0; c < 8; ++c) {
          const uint32_t idx = (uint32_t)(qx[c & 1] + qy[(c >> 1) & 1] + qz[c >> 2]);
          const float wgt = wxy[c & 3] * uzz[c >> 2];
          const float2 tv = *reinterpret_cast<const float2*>(tab + 2u * idx);
          f0 = fmaf(wgt, tv.x, f0);
          f1 = fmaf(wgt, tv.y, f1);
        }
      } else {
        const uint32_t hx[2] = {(uint32_t)ix, (uint32_t)ix + 1u};
        const uint32_t hy0 = (uint32_t)iy * P1;
        const uint32_t hy[2] = {hy0, hy0 + P1};
        const uint32_t hz0 = (uint32_t)iz * P2;
        const uint32_t hz[2] = {hz0, hz0 + P2};
#pragma unroll
        for (int c = 0; c < 8; ++c) {
          const uint32_t idx =
              (hx[c & 1] ^ hy[(c >> 1) & 1] ^ hz[c >> 2]) & (uint32_t)(TSIZE - 1);
          const float wgt = wxy[c & 3] * uzz[c >> 2];
          const float2 tv = *reinterpret_cast<const float2*>(tab + 2u * idx);
          f0 = fmaf(wgt, tv.x, f0);
          f1 = fmaf(wgt, tv.y, f1);
        }
      }
      v2h pk = {(_Float16)f0, (_Float16)f1};
      *reinterpret_cast<v2h*>(&sEnc[wv][m][2 * l]) = pk;  // 4B DS store
    }
    // Intra-wave LDS RAW: DS ops are in-order per wave; no block barrier needed.

    // ================= layer 1: [16x32] x [32x64] =================
    const _Float16* encRow = &sEnc[wv][m][0];
    v16h a1 = cat8(*reinterpret_cast<const v8h*>(encRow + k0),
                   *reinterpret_cast<const v8h*>(encRow + k0 + 16));
    v8f acc[4];
#pragma unroll
    for (int t = 0; t < 4; ++t) {
      v8f cz = {};
      v16h b = *reinterpret_cast<const v16h*>(&sB1[t][lane][0]);
      acc[t] = __builtin_amdgcn_wmma_f32_16x16x32_f16(false, a1, false, b,
                                                      (short)0, cz, false, false);
    }
#pragma unroll
    for (int t = 0; t < 4; ++t)
#pragma unroll
      for (int r = 0; r < 8; ++r)  // C/D layout: row = r + 8*hi16, col = 16t + m
        sH[wv][r + 8 * hi16][t * 16 + m] = (_Float16)softplus10(acc[t][r]);

    // ================= layer 2: [16x64] x [64x64] =================
    const _Float16* hRow = &sH[wv][m][0];
    {
      v16h a0 = cat8(*reinterpret_cast<const v8h*>(hRow + k0),
                     *reinterpret_cast<const v8h*>(hRow + k0 + 16));
      v16h a1b = cat8(*reinterpret_cast<const v8h*>(hRow + 32 + k0),
                      *reinterpret_cast<const v8h*>(hRow + 32 + k0 + 16));
#pragma unroll
      for (int t = 0; t < 4; ++t) {
        v8f cacc = {};
        v16h b0 = *reinterpret_cast<const v16h*>(&sB2[0][t][lane][0]);
        cacc = __builtin_amdgcn_wmma_f32_16x16x32_f16(false, a0, false, b0,
                                                      (short)0, cacc, false, false);
        v16h b1 = *reinterpret_cast<const v16h*>(&sB2[1][t][lane][0]);
        cacc = __builtin_amdgcn_wmma_f32_16x16x32_f16(false, a1b, false, b1,
                                                      (short)0, cacc, false, false);
        acc[t] = cacc;
      }
    }
#pragma unroll
    for (int t = 0; t < 4; ++t)
#pragma unroll
      for (int r = 0; r < 8; ++r)
        sH[wv][r + 8 * hi16][t * 16 + m] = (_Float16)softplus10(acc[t][r]);

    // ================= layer 3: [16x64] x [64x64] =================
    {
      v16h a0 = cat8(*reinterpret_cast<const v8h*>(hRow + k0),
                     *reinterpret_cast<const v8h*>(hRow + k0 + 16));
      v16h a1b = cat8(*reinterpret_cast<const v8h*>(hRow + 32 + k0),
                      *reinterpret_cast<const v8h*>(hRow + 32 + k0 + 16));
#pragma unroll
      for (int t = 0; t < 4; ++t) {
        v8f cacc = {};
        v16h b0 = *reinterpret_cast<const v16h*>(&sB3[0][t][lane][0]);
        cacc = __builtin_amdgcn_wmma_f32_16x16x32_f16(false, a0, false, b0,
                                                      (short)0, cacc, false, false);
        v16h b1 = *reinterpret_cast<const v16h*>(&sB3[1][t][lane][0]);
        cacc = __builtin_amdgcn_wmma_f32_16x16x32_f16(false, a1b, false, b1,
                                                      (short)0, cacc, false, false);
        acc[t] = cacc;
      }
    }
#pragma unroll
    for (int t = 0; t < 4; ++t)
#pragma unroll
      for (int r = 0; r < 8; ++r)
        sH[wv][r + 8 * hi16][t * 16 + m] = (_Float16)softplus10(acc[t][r]);

    // ===== layer 4: [16x64] x [64x1], vectorized LDS reads + xor-shuffle =====
    {
      const int koff = hi16 * 32;
      const v8h* hv = reinterpret_cast<const v8h*>(&sH[wv][m][koff]);
      const float4* wv4 = reinterpret_cast<const float4*>(&sW4[koff]);
      float sum = 0.0f;
#pragma unroll
      for (int ch = 0; ch < 4; ++ch) {
        const v8h hh = hv[ch];           // ds_load_b128
        const float4 wa = wv4[2 * ch];   // ds_load_b128
        const float4 wb = wv4[2 * ch + 1];
        sum = fmaf((float)hh[0], wa.x, sum);
        sum = fmaf((float)hh[1], wa.y, sum);
        sum = fmaf((float)hh[2], wa.z, sum);
        sum = fmaf((float)hh[3], wa.w, sum);
        sum = fmaf((float)hh[4], wb.x, sum);
        sum = fmaf((float)hh[5], wb.y, sum);
        sum = fmaf((float)hh[6], wb.z, sum);
        sum = fmaf((float)hh[7], wb.w, sum);
      }
      sum += __shfl_xor(sum, 16, 32);
      if (lane < 16) out[tile * 16 + m] = sum;
    }
  }
}

extern "C" void kernel_launch(void* const* d_in, const int* in_sizes, int n_in,
                              void* d_out, int out_size, void* d_ws,
                              size_t ws_size, hipStream_t stream) {
  const float* x     = (const float*)d_in[0];
  const float* table = (const float*)d_in[1];
  const float* w1    = (const float*)d_in[2];
  const float* w2    = (const float*)d_in[3];
  const float* w3    = (const float*)d_in[4];
  const float* w4    = (const float*)d_in[5];
  float* out = (float*)d_out;

  const int n_points = in_sizes[0] / 3;   // 2^20
  const int n_tiles = n_points / 16;      // 65536 full 16-point WMMA tiles
  // 2048 blocks x 8 waves = 16384 waves -> exactly 4 tiles per wave.
  dim3 grid(2048), block(256);
  hipLaunchKernelGGL(ngp_sdf_kernel, grid, block, 0, stream, x, table, w1, w2,
                     w3, w4, out, n_tiles);
}
